// EGCL_8203387535901
// MI455X (gfx1250) — matrix operand order repaired and checked by reading, not compiled, not run on hardware
//
#include <hip/hip_runtime.h>
#include <math.h>

typedef _Float16 h16;
typedef __attribute__((ext_vector_type(8)))  _Float16 v8h;
typedef __attribute__((ext_vector_type(16))) _Float16 v16h;
typedef __attribute__((ext_vector_type(8)))  float    v8f;

static constexpr int N_  = 2048;    // nodes
static constexpr int NN_ = 32768;   // edges
static constexpr int F_  = 256;     // feature dim
static constexpr int H_  = 256;     // hidden dim
static constexpr int KP  = 544;     // padded K for 517-wide edge-in / 520-wide node-in (17*32)

__device__ __forceinline__ float siluf(float x) { return x / (1.f + expf(-x)); }

// ---------------- zero fill (agg + coord accumulators each call) ----------------
__global__ void k_zero(float* __restrict__ p, int n) {
    int i = blockIdx.x * blockDim.x + threadIdx.x;
    if (i < n) p[i] = 0.f;
}

// ---------------- weight convert+transpose: Wt[n*Kp+k] = k<Kin ? W[k*256+n] : 0 ----------------
__global__ void k_wt(const float* __restrict__ W, h16* __restrict__ Wt, int Kin, int Kp) {
    int n = blockIdx.y;                       // 0..255 output column
    int k = blockIdx.x * 256 + threadIdx.x;   // K index
    if (k < Kp)
        Wt[(size_t)n * Kp + k] = (k < Kin) ? (h16)W[(size_t)k * 256 + n] : (h16)0.f;
}

// ---------------- build padded f16 edge MLP input [NN, KP] + diff/dist ----------------
__global__ void k_edge_in(const float* __restrict__ coords,
                          const float* __restrict__ feats,
                          const int*   __restrict__ edges,
                          const float* __restrict__ eattr,
                          h16*  __restrict__ mlp,
                          float* __restrict__ diff,
                          float* __restrict__ dist) {
    int e = blockIdx.x, t = threadIdx.x;
    int i = edges[2 * e + 0], j = edges[2 * e + 1];
    float dx = coords[3 * i + 0] - coords[3 * j + 0];
    float dy = coords[3 * i + 1] - coords[3 * j + 1];
    float dz = coords[3 * i + 2] - coords[3 * j + 2];
    float d2 = dx * dx + dy * dy + dz * dz;
    h16* row = mlp + (size_t)e * KP;
    row[t]      = (h16)feats[(size_t)i * F_ + t];
    row[F_ + t] = (h16)feats[(size_t)j * F_ + t];
    if (t < KP - 512) {                       // columns 512..543
        int c = 512 + t;
        float v = 0.f;
        if (c == 512) v = d2;                 // d_e^2
        else if (c < 517) v = eattr[(size_t)e * 4 + (c - 513)];
        row[c] = (h16)v;
    }
    if (t == 0) {
        dist[e] = sqrtf(d2);
        diff[3 * e + 0] = dx; diff[3 * e + 1] = dy; diff[3 * e + 2] = dz;
    }
}

// ---------------- epilogue helper: bias + optional SiLU, store f16/f32 ----------------
__device__ __forceinline__ void epi(v8f acc, int col, int mt, bool lo, float bv, int act,
                                    h16* __restrict__ outH, float* __restrict__ outF) {
#pragma unroll
    for (int r = 0; r < 8; ++r) {
        int row = mt * 16 + (lo ? r : r + 8);     // ISA C/D layout: VGPR r -> M=r (lanes 0-15), M=8+r (lanes 16-31)
        float v = acc[r] + bv;
        if (act) v = siluf(v);
        size_t idx = (size_t)row * 256 + col;
        if (outH) outH[idx] = (h16)v;
        if (outF) outF[idx] = v;
    }
}

// ---------------- WMMA GEMM: C[M,256] = act(A[M,K] @ Bt^T + bias) ----------------
// A row-major f16, K multiple of 32, 16B-aligned rows. Bt = B transposed: [256, K] f16.
// 4 waves/block; each wave computes a 32(M) x 64(N) strip = 2 M-tiles x 4 N-tiles.
// Per K-step: 4 A loads + 4 B loads feed 8 v_wmma_f32_16x16x32_f16 (B reused across M-tiles,
// A reused across N-tiles) -> 0.67 WMMA per b128 issue.
__global__ __launch_bounds__(128) void k_gemm(const h16* __restrict__ A,
                                              const h16* __restrict__ Bt,
                                              const float* __restrict__ bias,
                                              h16*  __restrict__ outH,
                                              float* __restrict__ outF,
                                              int M, int K, int act) {
    const int wave = threadIdx.x >> 5;
    const int lane = threadIdx.x & 31;
    const int mt0 = (blockIdx.x * 4 + wave) * 2;  // first 16-row tile index
    const int mt1 = mt0 + 1;
    const int n0 = blockIdx.y * 64;               // 64-col strip
    if (mt0 * 16 >= M) return;
    const bool lo = lane < 16;
    const int  ln = lane & 15;

    // A fragment (16x32 f16): lane holds row M=ln; lanes<16 K=0..15, lanes>=16 K=16..31 (matches B)
    const h16* arow0 = A  + (size_t)(mt0 * 16 + ln) * K + (lo ? 0 : 16);
    const h16* arow1 = A  + (size_t)(mt1 * 16 + ln) * K + (lo ? 0 : 16);
    // B fragment (32x16 f16): lane holds column N; lanes<16 K=0..15, lanes>=16 K=16..31 (contiguous in Bt rows)
    const h16* brow0 = Bt + (size_t)(n0 +  0 + ln) * K + (lo ? 0 : 16);
    const h16* brow1 = Bt + (size_t)(n0 + 16 + ln) * K + (lo ? 0 : 16);
    const h16* brow2 = Bt + (size_t)(n0 + 32 + ln) * K + (lo ? 0 : 16);
    const h16* brow3 = Bt + (size_t)(n0 + 48 + ln) * K + (lo ? 0 : 16);

    v8f acc00 = {}, acc01 = {}, acc02 = {}, acc03 = {};
    v8f acc10 = {}, acc11 = {}, acc12 = {}, acc13 = {};
    for (int k = 0; k < K; k += 32) {
        v8h a0lo = *(const v8h*)(arow0 + k);
        v8h a0hi = *(const v8h*)(arow0 + k + 8);
        v8h a1lo = *(const v8h*)(arow1 + k);
        v8h a1hi = *(const v8h*)(arow1 + k + 8);
        v16h a0, a1;
#pragma unroll
        for (int i = 0; i < 8; ++i) {
            a0[i] = a0lo[i]; a0[i + 8] = a0hi[i];
            a1[i] = a1lo[i]; a1[i + 8] = a1hi[i];
        }
        v16h b0 = *(const v16h*)(brow0 + k);
        v16h b1 = *(const v16h*)(brow1 + k);
        v16h b2 = *(const v16h*)(brow2 + k);
        v16h b3 = *(const v16h*)(brow3 + k);
        acc00 = __builtin_amdgcn_wmma_f32_16x16x32_f16(false, a0, false, b0, (short)0, acc00, false, false);
        acc01 = __builtin_amdgcn_wmma_f32_16x16x32_f16(false, a0, false, b1, (short)0, acc01, false, false);
        acc02 = __builtin_amdgcn_wmma_f32_16x16x32_f16(false, a0, false, b2, (short)0, acc02, false, false);
        acc03 = __builtin_amdgcn_wmma_f32_16x16x32_f16(false, a0, false, b3, (short)0, acc03, false, false);
        acc10 = __builtin_amdgcn_wmma_f32_16x16x32_f16(false, a1, false, b0, (short)0, acc10, false, false);
        acc11 = __builtin_amdgcn_wmma_f32_16x16x32_f16(false, a1, false, b1, (short)0, acc11, false, false);
        acc12 = __builtin_amdgcn_wmma_f32_16x16x32_f16(false, a1, false, b2, (short)0, acc12, false, false);
        acc13 = __builtin_amdgcn_wmma_f32_16x16x32_f16(false, a1, false, b3, (short)0, acc13, false, false);
    }
    epi(acc00, n0 +  0 + ln, mt0, lo, bias[n0 +  0 + ln], act, outH, outF);
    epi(acc01, n0 + 16 + ln, mt0, lo, bias[n0 + 16 + ln], act, outH, outF);
    epi(acc02, n0 + 32 + ln, mt0, lo, bias[n0 + 32 + ln], act, outH, outF);
    epi(acc03, n0 + 48 + ln, mt0, lo, bias[n0 + 48 + ln], act, outH, outF);
    epi(acc10, n0 +  0 + ln, mt1, lo, bias[n0 +  0 + ln], act, outH, outF);
    epi(acc11, n0 + 16 + ln, mt1, lo, bias[n0 + 16 + ln], act, outH, outF);
    epi(acc12, n0 + 32 + ln, mt1, lo, bias[n0 + 32 + ln], act, outH, outF);
    epi(acc13, n0 + 48 + ln, mt1, lo, bias[n0 + 48 + ln], act, outH, outF);
}

// ---------------- per-edge gate + scatter-add (agg, coord accum) ----------------
__global__ void k_edge_reduce(const h16* __restrict__ phie,   // silu(silu(in@eW1)@eW2) f16 [NN,H]
                              const h16* __restrict__ tc,     // silu(silu(in@cW1)@cW2) f16 [NN,H]
                              const float* __restrict__ iW, const float* __restrict__ ib,
                              const float* __restrict__ cW3,
                              const int*   __restrict__ edges,
                              const float* __restrict__ diff,
                              const float* __restrict__ dist,
                              float* __restrict__ agg,
                              float* __restrict__ cacc) {
    __shared__ float red[H_];
    __shared__ float bc[2];
    int e = blockIdx.x, t = threadIdx.x;
    float pe = (float)phie[(size_t)e * H_ + t];
    float tv = (float)tc  [(size_t)e * H_ + t];

    red[t] = pe * iW[t];                                  // phi_e . iW
    __syncthreads();
    for (int s = H_ / 2; s > 0; s >>= 1) { if (t < s) red[t] += red[t + s]; __syncthreads(); }
    if (t == 0) bc[0] = red[0];
    __syncthreads();
    red[t] = tv * cW3[t];                                 // t_c . cW3 -> phi_x
    __syncthreads();
    for (int s = H_ / 2; s > 0; s >>= 1) { if (t < s) red[t] += red[t + s]; __syncthreads(); }
    if (t == 0) bc[1] = red[0];
    __syncthreads();

    float infv = 1.f / (1.f + expf(-(bc[0] + ib[0])));    // sigmoid gate
    int recv = edges[2 * e + 0];
    atomicAdd(&agg[(size_t)recv * H_ + t], pe * infv);
    if (t < 3) {
        float w = 15.f * tanhf(bc[1]);                    // TANH_RANGE * tanh(phi_x)
        float d = dist[e];
        float ncd = diff[3 * e + t] / (fmaxf(d, 1e-5f) + 1.f);
        atomicAdd(&cacc[recv * 3 + t], w * ncd);
    }
}

// ---------------- build padded f16 node MLP input [N, KP] ----------------
__global__ void k_node_in(const float* __restrict__ feats,
                          const float* __restrict__ agg,
                          const float* __restrict__ nattr,
                          h16* __restrict__ nin) {
    int n = blockIdx.x, t = threadIdx.x;
    h16* row = nin + (size_t)n * KP;
    row[t]      = (h16)feats[(size_t)n * F_ + t];
    row[F_ + t] = (h16)agg  [(size_t)n * H_ + t];
    if (t < KP - 512) {
        int c = 512 + t;
        row[c] = (c < 520) ? (h16)nattr[(size_t)n * 8 + (c - 512)] : (h16)0.f;
    }
}

// ---------------- residual finalize ----------------
__global__ void k_finalize(const float* __restrict__ coords,
                           const float* __restrict__ feats,
                           const float* __restrict__ cacc,
                           const float* __restrict__ out2,
                           float* __restrict__ outC,
                           float* __restrict__ outF) {
    int n = blockIdx.x, t = threadIdx.x;
    outF[(size_t)n * F_ + t] = feats[(size_t)n * F_ + t] + out2[(size_t)n * F_ + t];
    if (t < 3) outC[n * 3 + t] = coords[n * 3 + t] + cacc[n * 3 + t];
}

extern "C" void kernel_launch(void* const* d_in, const int* in_sizes, int n_in,
                              void* d_out, int out_size, void* d_ws, size_t ws_size,
                              hipStream_t stream) {
    (void)in_sizes; (void)n_in; (void)out_size; (void)ws_size;
    const float* coords = (const float*)d_in[0];
    const float* feats  = (const float*)d_in[1];
    const int*   edges  = (const int*)d_in[2];
    // d_in[3] = dense `reduce` matrix: intentionally unused (scatter via edges instead)
    const float* nattr  = (const float*)d_in[4];
    const float* eattr  = (const float*)d_in[5];
    const float* eW1 = (const float*)d_in[6];  const float* eb1 = (const float*)d_in[7];
    const float* eW2 = (const float*)d_in[8];  const float* eb2 = (const float*)d_in[9];
    const float* cW1 = (const float*)d_in[10]; const float* cb1 = (const float*)d_in[11];
    const float* cW2 = (const float*)d_in[12]; const float* cb2 = (const float*)d_in[13];
    const float* cW3 = (const float*)d_in[14];
    const float* nW1 = (const float*)d_in[15]; const float* nb1 = (const float*)d_in[16];
    const float* nW2 = (const float*)d_in[17]; const float* nb2 = (const float*)d_in[18];
    const float* iW  = (const float*)d_in[19]; const float* ib  = (const float*)d_in[20];

    // -------- workspace layout (bytes, all 256-aligned; total ~78.5 MB) --------
    char* ws = (char*)d_ws;
    h16*  mlp   = (h16*)(ws + 0);              // [NN,KP] f16 : 35,651,584
    h16*  phieH = (h16*)(ws + 0);              // alias: phi_e f16 [NN,H] after GEMM1/2 consumed mlp
    h16*  tcH   = (h16*)(ws + 16777216);       // alias: t_c  f16 [NN,H]
    h16*  h1e   = (h16*)(ws + 35651584);       // [NN,H] f16
    h16*  h1c   = (h16*)(ws + 52428800);       // [NN,H] f16
    h16*  eW1t  = (h16*)(ws + 69206016);       // [256,KP]
    h16*  cW1t  = (h16*)(ws + 69206016 + 278528);
    h16*  nW1t  = (h16*)(ws + 69206016 + 2 * 278528);
    h16*  eW2t  = (h16*)(ws + 69206016 + 3 * 278528);              // [256,256]
    h16*  cW2t  = (h16*)(ws + 69206016 + 3 * 278528 + 131072);
    h16*  nW2t  = (h16*)(ws + 69206016 + 3 * 278528 + 2 * 131072);
    float* diff = (float*)(ws + 70434816);     // [NN,3]
    float* dist = (float*)(ws + 70828032);     // [NN]
    float* agg  = (float*)(ws + 70959104);     // [N,H]  (zeroed; atomic target)
    float* cacc = (float*)(ws + 73056256);     // [N,3]  (zeroed; atomic target; adjacent to agg)
    h16*  nin   = (h16*)(ws + 73080832);       // [N,KP] f16
    h16*  hn1   = (h16*)(ws + 75309056);       // [N,H] f16
    float* out2 = (float*)(ws + 76357632);     // [N,F] f32

    // 1) zero scatter accumulators (agg + cacc are contiguous)
    int nz = N_ * H_ + N_ * 3;
    k_zero<<<(nz + 255) / 256, 256, 0, stream>>>(agg, nz);

    // 2) f16 transposed weights
    dim3 wg1((KP + 255) / 256, 256);
    k_wt<<<wg1, 256, 0, stream>>>(eW1, eW1t, 517, KP);
    k_wt<<<wg1, 256, 0, stream>>>(cW1, cW1t, 517, KP);
    k_wt<<<wg1, 256, 0, stream>>>(nW1, nW1t, 520, KP);
    dim3 wg2(1, 256);
    k_wt<<<wg2, 256, 0, stream>>>(eW2, eW2t, 256, 256);
    k_wt<<<wg2, 256, 0, stream>>>(cW2, cW2t, 256, 256);
    k_wt<<<wg2, 256, 0, stream>>>(nW2, nW2t, 256, 256);

    // 3) gather edge inputs
    k_edge_in<<<NN_, 256, 0, stream>>>(coords, feats, edges, eattr, mlp, diff, dist);

    // 4) edge MLPs (WMMA); each block covers 128 rows (4 waves x 2 M-tiles)
    dim3 ge(NN_ / 128, 4);
    k_gemm<<<ge, 128, 0, stream>>>(mlp, eW1t, eb1, h1e, nullptr, NN_, KP,  1);
    k_gemm<<<ge, 128, 0, stream>>>(mlp, cW1t, cb1, h1c, nullptr, NN_, KP,  1);
    k_gemm<<<ge, 128, 0, stream>>>(h1e, eW2t, eb2, phieH, nullptr, NN_, 256, 1);
    k_gemm<<<ge, 128, 0, stream>>>(h1c, cW2t, cb2, tcH,   nullptr, NN_, 256, 1);

    // 5) gate + scatter
    k_edge_reduce<<<NN_, 256, 0, stream>>>(phieH, tcH, iW, ib, cW3, edges, diff, dist, agg, cacc);

    // 6) node MLP (WMMA) + residual
    k_node_in<<<N_, 256, 0, stream>>>(feats, agg, nattr, nin);
    dim3 gn(N_ / 128, 4);
    k_gemm<<<gn, 128, 0, stream>>>(nin, nW1t, nb1, hn1, nullptr, N_, KP,  1);
    k_gemm<<<gn, 128, 0, stream>>>(hn1, nW2t, nb2, nullptr, out2, N_, 256, 0);

    float* outC = (float*)d_out;
    float* outF = outC + N_ * 3;
    k_finalize<<<N_, 256, 0, stream>>>(coords, feats, cacc, out2, outC, outF);
}
